// OnlineReweightingLoss_6073083757069
// MI455X (gfx1250) — compile-verified, matrix operand order checked by reading, not run on hardware
//
#include <hip/hip_runtime.h>

// Problem constants (match reference)
#define N_SAMPLES (1 << 20)   // 1,048,576
#define C 128                 // classes
#define S 4                   // subgroups
#define NGROUPS (C * S)       // 512

#define LSE_BLOCKS 1024
#define LSE_THREADS 256
#define LSE_WAVES ((LSE_BLOCKS * LSE_THREADS) / 32)  // 8192

typedef __attribute__((ext_vector_type(16))) _Float16 v16h;
typedef __attribute__((ext_vector_type(8)))  float    v8f;

// ---------------------------------------------------------------------------
// Kernel 0: zero the 512-bin group-count histogram in workspace
// ---------------------------------------------------------------------------
__global__ void orl_init_kernel(unsigned* __restrict__ counts) {
    int i = threadIdx.x;
    if (i < NGROUPS) counts[i] = 0u;
}

// ---------------------------------------------------------------------------
// Kernel 1: histogram of group_id = target*S + subgroup (LDS then global,
// integer atomics -> deterministic)
// ---------------------------------------------------------------------------
__global__ void orl_hist_kernel(const int* __restrict__ tgt,
                                const int* __restrict__ sg,
                                unsigned* __restrict__ counts) {
    __shared__ unsigned h[NGROUPS];
    for (int i = threadIdx.x; i < NGROUPS; i += blockDim.x) h[i] = 0u;
    __syncthreads();
    int stride = gridDim.x * blockDim.x;
    for (int i = blockIdx.x * blockDim.x + threadIdx.x; i < N_SAMPLES; i += stride)
        atomicAdd(&h[tgt[i] * S + sg[i]], 1u);
    __syncthreads();
    for (int i = threadIdx.x; i < NGROUPS; i += blockDim.x) {
        unsigned c = h[i];
        if (c) atomicAdd(&counts[i], c);
    }
}

// ---------------------------------------------------------------------------
// Kernel 2: per-row LSE via WMMA row-sum, weighted accumulate per wave.
// One wave processes a 16x128 tile. Data layout per lane matches the ISA's
// 16-bit A-matrix 16x32 layout: lane L holds row r = L%16, half h = L>>4,
// and within each 32-wide K chunk c the column runs [32c+8h .. +7] and
// [32c+16+8h .. +7]. Sum-of-exp per row comes from 4 chained
// v_wmma_f32_16x16x32_f16 with an all-ones B matrix.
// ---------------------------------------------------------------------------
__global__ void orl_lse_kernel(const float* __restrict__ logits,
                               const int* __restrict__ tgt,
                               const int* __restrict__ sg,
                               const unsigned* __restrict__ counts,
                               float* __restrict__ partials) {
    const int lane   = threadIdx.x & 31;
    const int wave   = (blockIdx.x * blockDim.x + threadIdx.x) >> 5;
    const int nwaves = (gridDim.x * blockDim.x) >> 5;
    const int r      = lane & 15;   // row within tile
    const int h      = lane >> 4;   // half-wave (K sub-block)
    const int ntiles = N_SAMPLES / 16;

    v16h bones;
#pragma unroll
    for (int i = 0; i < 16; ++i) bones[i] = (_Float16)1.0f;

    float acc = 0.0f;

    for (int tile = wave; tile < ntiles; tile += nwaves) {
        const int row0 = tile << 4;
        const float* rowp = logits + (size_t)(row0 + r) * C;

        // prefetch next tile for this wave (global_prefetch_b8)
        if (tile + nwaves < ntiles)
            __builtin_prefetch(rowp + (size_t)nwaves * 16 * C, 0, 1);

        // --- load 16x128 tile: 64 floats/lane as 16x float4 (b128 loads) ---
        float4 v[16];
#pragma unroll
        for (int c = 0; c < 4; ++c) {
#pragma unroll
            for (int s2 = 0; s2 < 2; ++s2) {
                const float4* p = (const float4*)(rowp + 32 * c + 16 * s2 + 8 * h);
                v[c * 4 + s2 * 2 + 0] = p[0];
                v[c * 4 + s2 * 2 + 1] = p[1];
            }
        }

        // --- per-row max: local 64-wide max, then combine lane pair (r, r+16)
        float m = -3.402823466e38f;
#pragma unroll
        for (int k = 0; k < 16; ++k)
            m = fmaxf(m, fmaxf(fmaxf(v[k].x, v[k].y), fmaxf(v[k].z, v[k].w)));
        m = fmaxf(m, __shfl_xor(m, 16));

        // --- per-row target logit + inverse group count (tiny cached loads)
        int   t    = tgt[row0 + r];
        int   g    = t * S + sg[row0 + r];
        float xt   = logits[(size_t)(row0 + r) * C + t];
        float rinv = 1.0f / (float)counts[g];

        // --- exp(x - m) -> f16 A tiles; row-sum via 4 chained WMMAs ---------
        v8f dacc = {};
#pragma unroll
        for (int c = 0; c < 4; ++c) {
            v16h a;
#pragma unroll
            for (int q = 0; q < 4; ++q) {
                float4 e = v[c * 4 + q];
                a[q * 4 + 0] = (_Float16)__expf(e.x - m);
                a[q * 4 + 1] = (_Float16)__expf(e.y - m);
                a[q * 4 + 2] = (_Float16)__expf(e.z - m);
                a[q * 4 + 3] = (_Float16)__expf(e.w - m);
            }
            dacc = __builtin_amdgcn_wmma_f32_16x16x32_f16(
                false, a, false, bones, (short)0, dacc, false, false);
        }

        // --- accumulate (m - x_t)/cnt : one lane per row (lanes 0..15) ------
        if (lane < 16) acc += (m - xt) * rinv;

        // --- accumulate log(sumexp)/cnt : D layout gives lane<16 rows 0..7
        //     (d[j] = row j) and lane>=16 rows 8..15 (d[j] = row j+8).
        //     Only lanes 0 and 16 contribute (all columns of D identical).
        float sel   = ((lane & 15) == 0) ? 1.0f : 0.0f;
        int   rbase = h << 3;
#pragma unroll
        for (int j = 0; j < 8; ++j) {
            float cj = __shfl(rinv, rbase + j);  // rinv of row rbase+j
            acc += sel * __logf(fmaxf(dacc[j], 1e-30f)) * cj;
        }
    }

    // wave-level reduction; one partial per wave (deterministic final pass)
#pragma unroll
    for (int off = 16; off >= 1; off >>= 1) acc += __shfl_xor(acc, off);
    if (lane == 0) partials[wave] = acc;
}

// ---------------------------------------------------------------------------
// Kernel 3: deterministic fixed-order reduction of per-wave partials
// ---------------------------------------------------------------------------
__global__ void orl_reduce_kernel(const float* __restrict__ partials,
                                  float* __restrict__ out) {
    __shared__ float sdata[256];
    float a = 0.0f;
    for (int i = threadIdx.x; i < LSE_WAVES; i += 256) a += partials[i];
    sdata[threadIdx.x] = a;
    __syncthreads();
    for (int w = 128; w > 0; w >>= 1) {
        if (threadIdx.x < w) sdata[threadIdx.x] += sdata[threadIdx.x + w];
        __syncthreads();
    }
    if (threadIdx.x == 0) out[0] = sdata[0];
}

// ---------------------------------------------------------------------------
extern "C" void kernel_launch(void* const* d_in, const int* in_sizes, int n_in,
                              void* d_out, int out_size, void* d_ws, size_t ws_size,
                              hipStream_t stream) {
    const float* logits = (const float*)d_in[0];
    const int*   tgt    = (const int*)d_in[1];
    const int*   sg     = (const int*)d_in[2];
    float*       out    = (float*)d_out;

    // workspace layout: [0, 2048) group counts (u32); [2048, ...) wave partials
    unsigned* counts   = (unsigned*)d_ws;
    float*    partials = (float*)((char*)d_ws + NGROUPS * sizeof(unsigned));

    orl_init_kernel<<<1, NGROUPS, 0, stream>>>(counts);
    orl_hist_kernel<<<512, 256, 0, stream>>>(tgt, sg, counts);
    orl_lse_kernel<<<LSE_BLOCKS, LSE_THREADS, 0, stream>>>(logits, tgt, sg,
                                                           counts, partials);
    orl_reduce_kernel<<<1, 256, 0, stream>>>(partials, out);
}
